// SelfAttention3D_18356690223488
// MI455X (gfx1250) — matrix operand order repaired and checked by reading, not compiled, not run on hardware
//
#include <hip/hip_runtime.h>
#include <hip/hip_bf16.h>

typedef _Float16 f16;
typedef __attribute__((ext_vector_type(16))) _Float16 v16h;
typedef __attribute__((ext_vector_type(8)))  float    v8f;

constexpr int NB   = 32;          // batch items (bs * t)
constexpr int TT   = 576;         // tokens per item (24*24)
constexpr int E    = 256;
constexpr int CH   = 64;
constexpr int NH   = 4;
constexpr int DH   = 64;
constexpr int MTOT = NB * TT;     // 18432 (= 144 * 128)
constexpr float SCALE = 0.125f;   // 1/sqrt(Dh)

__device__ inline v8f vzero8() {
  v8f z;
#pragma unroll
  for (int i = 0; i < 8; ++i) z[i] = 0.0f;
  return z;
}

__device__ inline v8f wmma16(v16h a, v16h b, v8f c) {
  // D = A(16x32 f16) * B(32x16 f16) + C(16x16 f32)
  return __builtin_amdgcn_wmma_f32_16x16x32_f16(false, a, false, b, (short)0, c,
                                                false, false);
}

union frag_cast { uint4 u[2]; v16h v; };

// A fragment 16x32 (M x K) from row-major source, leading dim ld (halves).
// lane: m = lane%16, g = lane/16; halves j -> K = g*8+j ; 8+j -> K = 16+g*8+j.
__device__ inline v16h load_a_frag(const f16* __restrict__ base, int ld) {
  const int lane = threadIdx.x & 31;
  const int m = lane & 15, g = lane >> 4;
  frag_cast t;
  t.u[0] = *(const uint4*)(base + m * ld + g * 8);
  t.u[1] = *(const uint4*)(base + m * ld + 16 + g * 8);
  return t.v;
}

// B fragment 32x16 (K x N): lane: n = lane%16, g = lane/16; halves j -> K = g*8+j ; 8+j -> K = 16+g*8+j.
__device__ inline v16h load_b_frag(const f16* __restrict__ base, int strideHalves) {
  const int lane = threadIdx.x & 31;
  const int n = lane & 15, g = lane >> 4;
  v16h t;
#pragma unroll
  for (int j = 0; j < 8; ++j) {
    t[j]     = base[(size_t)(g * 8 + j) * strideHalves + n];
    t[8 + j] = base[(size_t)(16 + g * 8 + j) * strideHalves + n];
  }
  return t;
}

// ---------------------------------------------------------------------------
// f16 WMMA GEMM: C[M,N] = A[M,K] * B[K,N]. M % 128 == 0, N % 128 == 0,
// K % 32 == 0 (caller pads B). Block: 256 threads = 8 waves as 4(M) x 2(N);
// block tile 128x128, wave tile 32x64 (2x4 C-fragments), 32-wide K tiles.
// Software pipelined: tile k+1 is in flight (global_load_b128 -> regs) while
// the 8 WMMAs of tile k run out of LDS.
// EPI: 0 f32 = acc+bias            (conv-in)
//      1 f16 = acc                 (q / xav projections)
//      2 f32 = acc+bias+resid      (attn out + residual)
//      3 f16 = gelu(acc+bias)      (mlp up)
//      4 f16 = acc+bias+resid      (mlp down + residual)
//      5 scatter f32 = acc+bias to d_out in [bs,ch,t,h,w] order, col < CH only
// ---------------------------------------------------------------------------
template <int EPI>
__global__ __launch_bounds__(256) void k_gemm(
    const f16* __restrict__ A, const f16* __restrict__ B,
    int M, int N, int K,
    const float* __restrict__ bias, const float* __restrict__ resid,
    float* __restrict__ outF, f16* __restrict__ outH)
{
  __shared__ f16 sA[128 * 32];   // 8 KB
  __shared__ f16 sB[32 * 128];   // 8 KB
  const int bm = blockIdx.x * 128;
  const int bn = blockIdx.y * 128;
  const int tid  = threadIdx.x;
  const int lane = tid & 31;
  const int wave = tid >> 5;
  const int wm = (wave & 3) * 32;   // wave row base in tile
  const int wn = (wave >> 2) * 64;  // wave col base in tile

  v8f acc[2][4];
#pragma unroll
  for (int i = 0; i < 2; ++i)
#pragma unroll
    for (int j = 0; j < 4; ++j) acc[i][j] = vzero8();

  uint4 ra[2], rb[2];
  auto gload = [&](int k0) {
#pragma unroll
    for (int l = 0; l < 2; ++l) {
      int c = tid + l * 256;
      ra[l] = *(const uint4*)(A + (size_t)(bm + (c >> 2)) * K + k0 + (c & 3) * 8);
      rb[l] = *(const uint4*)(B + (size_t)(k0 + (c >> 4)) * N + bn + (c & 15) * 8);
    }
  };
  auto sstore = [&]() {
#pragma unroll
    for (int l = 0; l < 2; ++l) {
      int c = tid + l * 256;
      *(uint4*)(sA + (c >> 2) * 32 + (c & 3) * 8)  = ra[l];
      *(uint4*)(sB + (c >> 4) * 128 + (c & 15) * 8) = rb[l];
    }
  };

  gload(0);
  for (int k0 = 0; k0 < K; k0 += 32) {
    sstore();
    __syncthreads();
    if (k0 + 32 < K) gload(k0 + 32);   // overlap next-tile loads with WMMAs

    v16h af[2], bf[4];
#pragma unroll
    for (int fm = 0; fm < 2; ++fm) af[fm] = load_a_frag(sA + (wm + fm * 16) * 32, 32);
#pragma unroll
    for (int fn = 0; fn < 4; ++fn) bf[fn] = load_b_frag(sB + wn + fn * 16, 128);
#pragma unroll
    for (int fm = 0; fm < 2; ++fm)
#pragma unroll
      for (int fn = 0; fn < 4; ++fn) acc[fm][fn] = wmma16(af[fm], bf[fn], acc[fm][fn]);
    __syncthreads();
  }

  // epilogue: C layout: row = r + 8*(lane/16), col = lane%16
  const int lm = lane & 15, lg = lane >> 4;
#pragma unroll
  for (int fm = 0; fm < 2; ++fm)
#pragma unroll
    for (int fn = 0; fn < 4; ++fn)
#pragma unroll
      for (int r = 0; r < 8; ++r) {
        int row = bm + wm + fm * 16 + r + lg * 8;
        int col = bn + wn + fn * 16 + lm;
        float v = acc[fm][fn][r];
        size_t o = (size_t)row * N + col;
        if constexpr (EPI == 0) {
          outF[o] = v + bias[col];
        } else if constexpr (EPI == 1) {
          outH[o] = (f16)v;
        } else if constexpr (EPI == 2) {
          outF[o] = v + bias[col] + resid[o];
        } else if constexpr (EPI == 3) {
          float xx = v + bias[col];
          outH[o] = (f16)(0.5f * xx * (1.0f + erff(xx * 0.70710678118f)));
        } else if constexpr (EPI == 4) {
          outH[o] = (f16)(v + bias[col] + resid[o]);
        } else {  // EPI == 5: row = (b*16+t)*576 + s ; out[b, col, t, s]
          if (col < CH) {
            int b  = row / (16 * TT);
            int rr = row - b * (16 * TT);
            int t  = rr / TT;
            int s  = rr - t * TT;
            outF[((size_t)(b * CH + col) * 16 + t) * TT + s] = v + bias[col];
          }
        }
      }
}

// ---------------------------------------------------------------------------
// Flash-style L2 attention (tied q=k). One wave per (item, head, 16-row tile).
// logits = (2*q_t.q_s - |q_t|^2 - |q_s|^2) * scale; online softmax over S in
// 64-chunks; O = softmax(L) @ XAV_head. Q^T is pre-transposed so every WMMA
// B-fragment k-row is a contiguous 32-B run across 16 lanes.
// ---------------------------------------------------------------------------
__global__ __launch_bounds__(256) void k_attn(
    const f16* __restrict__ Q,    // [M, E] row-major
    const f16* __restrict__ QT,   // [NB, NH, DH, TT]
    const float* __restrict__ SQ, // [M, NH]
    const f16* __restrict__ XAV,  // [M, E]
    f16* __restrict__ PXAV)       // [M, E]
{
  __shared__ f16 Pst[8][16 * 64];   // per-wave P staging (C-layout -> A-layout)
  const int wave = threadIdx.x >> 5;
  const int lane = threadIdx.x & 31;
  const int lm = lane & 15, lg = lane >> 4;

  int task = blockIdx.x * 8 + wave;                 // 4608 tasks = 576 blocks * 8
  int n   = task / (NH * (TT / 16));
  int rem = task % (NH * (TT / 16));
  int h   = rem / (TT / 16);
  int t0  = (rem % (TT / 16)) * 16;
  const size_t rowOff = (size_t)n * TT;
  const f16* Qh  = Q + rowOff * E + h * DH;
  const f16* QTh = QT + ((size_t)(n * NH + h)) * DH * TT;   // [DH][TT]

  // A-fragments of the 16 query rows (16 x 64, two K-chunks of 32)
  v16h aq[2];
#pragma unroll
  for (int kc = 0; kc < 2; ++kc)
    aq[kc] = load_a_frag(Qh + (size_t)t0 * E + kc * 32, E);

  float sqr[8], mrow[8], lsum[8];
#pragma unroll
  for (int r = 0; r < 8; ++r) {
    sqr[r]  = SQ[(rowOff + t0 + r + lg * 8) * NH + h];
    mrow[r] = -3.0e38f;
    lsum[r] = 0.0f;
  }
  v8f o[4];
#pragma unroll
  for (int dt = 0; dt < 4; ++dt) o[dt] = vzero8();

  for (int s0 = 0; s0 < TT; s0 += 64) {
    // ---- logits chunk 16 x 64 via WMMA (B from QT: lane = K feature row)
    v8f cf[4];
#pragma unroll
    for (int ct = 0; ct < 4; ++ct) {
      v16h b0 = load_b_frag(QTh + s0 + ct * 16, TT);
      v16h b1 = load_b_frag(QTh + (size_t)32 * TT + s0 + ct * 16, TT);
      v8f z = vzero8();
      z = wmma16(aq[0], b0, z);
      z = wmma16(aq[1], b1, z);
      float sqc = SQ[(rowOff + s0 + ct * 16 + lm) * NH + h];
#pragma unroll
      for (int r = 0; r < 8; ++r)
        cf[ct][r] = (2.0f * z[r] - sqr[r] - sqc) * SCALE;
    }
    // ---- online softmax (row reductions across the 16 lanes of a half)
#pragma unroll
    for (int r = 0; r < 8; ++r) {
      float mx = fmaxf(fmaxf(cf[0][r], cf[1][r]), fmaxf(cf[2][r], cf[3][r]));
      for (int off = 1; off < 16; off <<= 1) mx = fmaxf(mx, __shfl_xor(mx, off, 32));
      float mn = fmaxf(mrow[r], mx);
      float al = __expf(mrow[r] - mn);
      mrow[r] = mn;
      float ps = 0.0f;
#pragma unroll
      for (int ct = 0; ct < 4; ++ct) {
        float p = __expf(cf[ct][r] - mn);
        cf[ct][r] = p;
        ps += p;
      }
      for (int off = 1; off < 16; off <<= 1) ps += __shfl_xor(ps, off, 32);
      lsum[r] = lsum[r] * al + ps;
#pragma unroll
      for (int dt = 0; dt < 4; ++dt) o[dt][r] *= al;
    }
    // ---- stage P through LDS to convert C-layout -> A-layout
#pragma unroll
    for (int ct = 0; ct < 4; ++ct)
#pragma unroll
      for (int r = 0; r < 8; ++r)
        Pst[wave][(r + lg * 8) * 64 + ct * 16 + lm] = (f16)cf[ct][r];
    __syncthreads();   // uniform: all 8 waves run identical trip counts
    v16h ap[2];
#pragma unroll
    for (int kc = 0; kc < 2; ++kc)
      ap[kc] = load_a_frag(&Pst[wave][kc * 32], 64);
    __syncthreads();
    // ---- O += P @ XAV_head   (B frag: lane = s row, contiguous 16 d-cols)
#pragma unroll
    for (int kc = 0; kc < 2; ++kc) {
      const f16* xs = XAV + (rowOff + s0 + kc * 32) * E + h * DH;
#pragma unroll
      for (int dt = 0; dt < 4; ++dt)
        o[dt] = wmma16(ap[kc], load_b_frag(xs + dt * 16, E), o[dt]);
    }
  }
  // ---- normalize and store PXAV (f16)
#pragma unroll
  for (int dt = 0; dt < 4; ++dt)
#pragma unroll
    for (int r = 0; r < 8; ++r) {
      float v = o[dt][r] / lsum[r];
      PXAV[(rowOff + t0 + r + lg * 8) * E + h * DH + dt * 16 + lm] = (f16)v;
    }
}

// ---------------------------------------------------------------------------
// LayerNorm: one wave per 256-wide row, 8 rows / block, f32 in -> f16 out.
// ---------------------------------------------------------------------------
__global__ __launch_bounds__(256) void k_ln(
    const float* __restrict__ X, const float* __restrict__ g,
    const float* __restrict__ bta, f16* __restrict__ U)
{
  int row  = blockIdx.x * 8 + (threadIdx.x >> 5);
  int lane = threadIdx.x & 31;
  const float* x = X + (size_t)row * E;
  float v[8], s = 0.0f, s2 = 0.0f;
#pragma unroll
  for (int j = 0; j < 8; ++j) {
    v[j] = x[lane + 32 * j];
    s += v[j];
    s2 += v[j] * v[j];
  }
  for (int off = 16; off > 0; off >>= 1) {
    s  += __shfl_xor(s, off, 32);
    s2 += __shfl_xor(s2, off, 32);
  }
  float mu  = s * (1.0f / E);
  float var = s2 * (1.0f / E) - mu * mu;
  float rs  = rsqrtf(var + 1e-5f);
#pragma unroll
  for (int j = 0; j < 8; ++j) {
    int c = lane + 32 * j;
    U[(size_t)row * E + c] = (f16)((v[j] - mu) * rs * g[c] + bta[c]);
  }
}

// ||q||^2 per (token, head)
__global__ void k_sq(const f16* __restrict__ Q, float* __restrict__ SQ)
{
  int idx = blockIdx.x * blockDim.x + threadIdx.x;
  if (idx >= MTOT * NH) return;
  int m = idx >> 2, h = idx & 3;
  const f16* q = Q + (size_t)m * E + h * DH;
  float s = 0.0f;
#pragma unroll
  for (int d = 0; d < DH; ++d) {
    float t = (float)q[d];
    s += t * t;
  }
  SQ[idx] = s;
}

// Q [M,E] -> QT [NB, NH, DH, TT] (coalesced writes)
__global__ void k_qt(const f16* __restrict__ Q, f16* __restrict__ QT)
{
  int idx = blockIdx.x * blockDim.x + threadIdx.x;
  if (idx >= NB * NH * DH * TT) return;
  int s  = idx % TT;
  int r  = idx / TT;
  int d  = r % DH;
  int r2 = r / DH;
  int h  = r2 % NH;
  int n  = r2 / NH;
  QT[idx] = Q[((size_t)n * TT + s) * E + h * DH + d];
}

// Precompute per-head value-path weight: W_h = scale * Qw_h (Qw_h^T V_h),
// packed as Wxav[256, 256] f16 (head h in column block h*64..h*64+63).
__global__ __launch_bounds__(256) void k_wxav(
    const float* __restrict__ qw, const float* __restrict__ vw,
    f16* __restrict__ Wxav)
{
  __shared__ float QtV[64 * 64];   // 16 KB
  const int h = blockIdx.x;
  const int tid = threadIdx.x;
#pragma unroll 4
  for (int p = 0; p < 16; ++p) {
    int e = tid + 256 * p;
    int d1 = e >> 6, d2 = e & 63;
    float s = 0.0f;
    for (int m = 0; m < E; ++m)
      s += qw[m * E + h * DH + d1] * vw[m * E + h * DH + d2];
    QtV[e] = s;
  }
  __syncthreads();
  for (int p = 0; p < 64; ++p) {
    int e = tid + 256 * p;
    int m = e >> 6, d2 = e & 63;
    float s = 0.0f;
#pragma unroll 8
    for (int d1 = 0; d1 < DH; ++d1)
      s += qw[m * E + h * DH + d1] * QtV[d1 * 64 + d2];
    Wxav[(size_t)m * E + h * DH + d2] = (f16)(s * SCALE);
  }
}

// x [bs,ch,t,h,w] -> token matrix Xtok[(b*16+t)*576 + s, ch] in f16
__global__ void k_pack_x(const float* __restrict__ x, f16* __restrict__ Xtok)
{
  int idx = blockIdx.x * blockDim.x + threadIdx.x;
  if (idx >= MTOT * CH) return;
  int m = idx / CH, c = idx - m * CH;
  int b  = m / (16 * TT);
  int rr = m - b * (16 * TT);
  int t  = rr / TT;
  int s  = rr - t * TT;
  Xtok[idx] = (f16)x[((size_t)(b * CH + c) * 16 + t) * TT + s];
}

__global__ void k_cvt(const float* __restrict__ s, f16* __restrict__ d, int n)
{
  int i = blockIdx.x * blockDim.x + threadIdx.x;
  if (i < n) d[i] = (f16)s[i];
}

// w_out [E, CH] f32 -> zero-padded [E, 128] f16 so the GEMM never guards N
__global__ void k_pad_wout(const float* __restrict__ s, f16* __restrict__ d)
{
  int i = blockIdx.x * blockDim.x + threadIdx.x;
  if (i >= E * 128) return;
  int r = i >> 7, c = i & 127;
  d[i] = (c < CH) ? (f16)s[r * CH + c] : (f16)0.f;
}

// ---------------------------------------------------------------------------
extern "C" void kernel_launch(void* const* d_in, const int* in_sizes, int n_in,
                              void* d_out, int out_size, void* d_ws, size_t ws_size,
                              hipStream_t stream)
{
  const float* x     = (const float*)d_in[0];
  const float* w_in  = (const float*)d_in[1];
  const float* b_in  = (const float*)d_in[2];
  const float* ln1_g = (const float*)d_in[3];
  const float* ln1_b = (const float*)d_in[4];
  const float* q_w   = (const float*)d_in[5];   // [E,H,Dh] == row-major [256,256]
  const float* v_w   = (const float*)d_in[6];
  const float* w_ao  = (const float*)d_in[7];
  const float* b_ao  = (const float*)d_in[8];
  const float* ln2_g = (const float*)d_in[9];
  const float* ln2_b = (const float*)d_in[10];
  const float* w1    = (const float*)d_in[11];
  const float* b1    = (const float*)d_in[12];
  const float* w2    = (const float*)d_in[13];
  const float* b2    = (const float*)d_in[14];
  const float* w_out = (const float*)d_in[15];
  const float* b_out = (const float*)d_in[16];
  float* out = (float*)d_out;

  char* ws = (char*)d_ws;
  size_t off = 0;
  auto take = [&](size_t bytes) -> char* {
    char* p = ws + off;
    off = (off + bytes + 255) & ~(size_t)255;
    return p;
  };
  f16*   Xtok  = (f16*)take((size_t)MTOT * CH * 2);
  f16*   Win   = (f16*)take((size_t)CH * E * 2);
  f16*   Wq    = (f16*)take((size_t)E * E * 2);
  f16*   Wxav  = (f16*)take((size_t)E * E * 2);
  f16*   Wao   = (f16*)take((size_t)E * E * 2);
  f16*   W1    = (f16*)take((size_t)E * E * 2);
  f16*   W2    = (f16*)take((size_t)E * E * 2);
  f16*   WoutP = (f16*)take((size_t)E * 128 * 2);      // zero-padded conv-out W
  float* Afeat = (float*)take((size_t)MTOT * E * 4);   // conv-in out (residual 1)
  f16*   U     = (f16*)take((size_t)MTOT * E * 2);     // LN1 out; reused for LN2 out
  f16*   Qb    = (f16*)take((size_t)MTOT * E * 2);     // q proj; reused for MLP hidden
  f16*   QT    = (f16*)take((size_t)MTOT * E * 2);     // q transposed per (n,h)
  float* SQb   = (float*)take((size_t)MTOT * NH * 4);
  f16*   XAVb  = (f16*)take((size_t)MTOT * E * 2);     // xav; reused for MLP out
  f16*   PXAV  = (f16*)take((size_t)MTOT * E * 2);
  float* OUT1  = (float*)take((size_t)MTOT * E * 4);   // residual 2

  // pack input + weights to f16; precompute value-path weight
  k_pack_x<<<(MTOT * CH + 255) / 256, 256, 0, stream>>>(x, Xtok);
  k_cvt<<<(CH * E + 255) / 256, 256, 0, stream>>>(w_in, Win, CH * E);
  k_cvt<<<(E * E + 255) / 256, 256, 0, stream>>>(q_w, Wq, E * E);
  k_cvt<<<(E * E + 255) / 256, 256, 0, stream>>>(w_ao, Wao, E * E);
  k_cvt<<<(E * E + 255) / 256, 256, 0, stream>>>(w1, W1, E * E);
  k_cvt<<<(E * E + 255) / 256, 256, 0, stream>>>(w2, W2, E * E);
  k_pad_wout<<<(E * 128 + 255) / 256, 256, 0, stream>>>(w_out, WoutP);
  k_wxav<<<NH, 256, 0, stream>>>(q_w, v_w, Wxav);

  dim3 g2(MTOT / 128, 2), g1(MTOT / 128, 1);
  // a = conv_in(x) + b_in                                   (f32, residual 1)
  k_gemm<0><<<g2, 256, 0, stream>>>(Xtok, Win, MTOT, E, CH, b_in, nullptr, Afeat, nullptr);
  // u = LN1(a)
  k_ln<<<MTOT / 8, 256, 0, stream>>>(Afeat, ln1_g, ln1_b, U);
  // q = u @ Wq ; q^T ; |q|^2 ; xav = u @ Wxav
  k_gemm<1><<<g2, 256, 0, stream>>>(U, Wq, MTOT, E, E, nullptr, nullptr, nullptr, Qb);
  k_qt<<<(MTOT * E + 255) / 256, 256, 0, stream>>>(Qb, QT);
  k_sq<<<(MTOT * NH + 255) / 256, 256, 0, stream>>>(Qb, SQb);
  k_gemm<1><<<g2, 256, 0, stream>>>(U, Wxav, MTOT, E, E, nullptr, nullptr, nullptr, XAVb);
  // pxav = softmax(-L2) @ xav  (flash, per head)
  k_attn<<<(NB * NH * (TT / 16)) / 8, 256, 0, stream>>>(Qb, QT, SQb, XAVb, PXAV);
  // out1 = pxav @ w_ao + b_ao + a                           (f32, residual 2)
  k_gemm<2><<<g2, 256, 0, stream>>>(PXAV, Wao, MTOT, E, E, b_ao, Afeat, OUT1, nullptr);
  // MLP: gelu(LN2(out1) @ w1 + b1) @ w2 + b2 + out1
  k_ln<<<MTOT / 8, 256, 0, stream>>>(OUT1, ln2_g, ln2_b, U);
  k_gemm<3><<<g2, 256, 0, stream>>>(U, W1, MTOT, E, E, b1, nullptr, nullptr, Qb);
  k_gemm<4><<<g2, 256, 0, stream>>>(Qb, W2, MTOT, E, E, b2, OUT1, nullptr, XAVb);
  // conv_out (N padded to 128, cols >= 64 dropped) + scatter to [bs,ch,t,h,w]
  k_gemm<5><<<g1, 256, 0, stream>>>(XAVb, WoutP, MTOT, 128, E, b_out, nullptr, out, nullptr);

  (void)in_sizes; (void)n_in; (void)out_size; (void)ws_size; (void)off;
}
